// HQDiTLinear_37477884625348
// MI455X (gfx1250) — compile-verified, matrix operand order checked by reading, not run on hardware
//
#include <hip/hip_runtime.h>

typedef __attribute__((ext_vector_type(16))) _Float16 v16h;
typedef __attribute__((ext_vector_type(8)))  _Float16 v8h;
typedef __attribute__((ext_vector_type(8)))  float    v8f;

// ---------------------------------------------------------------------------
// FP4 fake-quant helpers (round-to-nearest, ties to LOWER level == jnp.argmin
// first-match semantics).
// NVFP4 levels: 0, .5, 1, 1.5, 2, 3, 4, 6   midpoints: .25 .75 1.25 1.75 2.5 3.5 5
// ---------------------------------------------------------------------------
__device__ __forceinline__ float q_nvfp4(float a) {
    return (a <= 0.25f) ? 0.0f :
           (a <= 0.75f) ? 0.5f :
           (a <= 1.25f) ? 1.0f :
           (a <= 1.75f) ? 1.5f :
           (a <= 2.50f) ? 2.0f :
           (a <= 3.50f) ? 3.0f :
           (a <= 5.00f) ? 4.0f : 6.0f;
}
// E1M2 levels: 0 .. 3.5 step 0.5 (uniform). round-half-down = ceil(2a - .5).
__device__ __forceinline__ float q_e1m2(float a) {
    float q = ceilf(2.0f * a - 0.5f);
    q = fminf(fmaxf(q, 0.0f), 7.0f);
    return 0.5f * q;
}

// ---------------------------------------------------------------------------
// Fast Walsh-Hadamard over one 64-block. Thread t holds 16 contiguous
// elements (its quant group); 4 lanes (t%4) form one 64-block.
// Stages 1..8 intra-thread, stages 16/32 via wave32 shfl_xor. Result *= 1/8.
// ---------------------------------------------------------------------------
__device__ __forceinline__ void fwht64(float v[16], int t) {
#pragma unroll
    for (int s = 1; s < 16; s <<= 1) {
#pragma unroll
        for (int i = 0; i < 16; i++) {
            if (!(i & s)) {
                float a = v[i], b = v[i | s];
                v[i] = a + b;
                v[i | s] = a - b;
            }
        }
    }
#pragma unroll
    for (int m = 1; m <= 2; m <<= 1) {
        bool up = (t & m) != 0;
#pragma unroll
        for (int i = 0; i < 16; i++) {
            float o = __shfl_xor(v[i], m, 32);
            v[i] = up ? (o - v[i]) : (v[i] + o);
        }
    }
#pragma unroll
    for (int i = 0; i < 16; i++) v[i] *= 0.125f;
}

// ---------------------------------------------------------------------------
// Activation path: rotate + NVFP4 fake-quant -> f16. One block per row.
// ---------------------------------------------------------------------------
__global__ __launch_bounds__(256) void rotq_act(const float* __restrict__ x,
                                                const float* __restrict__ signs,
                                                _Float16* __restrict__ xq, int K) {
    const int t = threadIdx.x;
    const size_t base = (size_t)blockIdx.x * K + t * 16;
    float v[16];
#pragma unroll
    for (int i = 0; i < 16; i++) v[i] = x[base + i] * signs[t * 16 + i];
    fwht64(v, t);

    float amax = 0.0f;
#pragma unroll
    for (int i = 0; i < 16; i++) amax = fmaxf(amax, fabsf(v[i]));
    amax = fmaxf(amax, 1e-12f);
    const float scale = amax * (1.0f / 6.0f);
#pragma unroll
    for (int i = 0; i < 16; i++) {
        float q = q_nvfp4(fabsf(v[i]) / scale) * scale;
        xq[base + i] = (_Float16)copysignf(q, v[i]);
    }
}

// ---------------------------------------------------------------------------
// Weight path: rotate, quantize both formats, per-row MSE format select,
// write selected fake-quant as f16. One block per output channel.
// ---------------------------------------------------------------------------
__global__ __launch_bounds__(256) void rotq_wgt(const float* __restrict__ w,
                                                const float* __restrict__ signs,
                                                _Float16* __restrict__ wq, int K) {
    const int t = threadIdx.x;
    const size_t base = (size_t)blockIdx.x * K + t * 16;
    float v[16];
#pragma unroll
    for (int i = 0; i < 16; i++) v[i] = w[base + i] * signs[t * 16 + i];
    fwht64(v, t);

    float amax = 0.0f;
#pragma unroll
    for (int i = 0; i < 16; i++) amax = fmaxf(amax, fabsf(v[i]));
    amax = fmaxf(amax, 1e-12f);
    const float s_nv = amax * (1.0f / 6.0f);
    const float s_e1 = amax * (1.0f / 3.5f);

    float e_nv = 0.0f, e_e1 = 0.0f;
#pragma unroll
    for (int i = 0; i < 16; i++) {
        float av = fabsf(v[i]);
        float qn = copysignf(q_nvfp4(av / s_nv) * s_nv, v[i]);
        float qe = copysignf(q_e1m2(av / s_e1) * s_e1, v[i]);
        float dn = qn - v[i], de = qe - v[i];
        e_nv += dn * dn;
        e_e1 += de * de;
    }
    __shared__ float red0[8], red1[8];
    __shared__ int sel;
#pragma unroll
    for (int off = 16; off > 0; off >>= 1) {
        e_nv += __shfl_down(e_nv, off, 32);
        e_e1 += __shfl_down(e_e1, off, 32);
    }
    if ((t & 31) == 0) { red0[t >> 5] = e_nv; red1[t >> 5] = e_e1; }
    __syncthreads();
    if (t == 0) {
        float s0 = 0.0f, s1 = 0.0f;
#pragma unroll
        for (int i = 0; i < 8; i++) { s0 += red0[i]; s1 += red1[i]; }
        sel = (s1 < s0) ? 1 : 0;
    }
    __syncthreads();
    const bool use_e1 = (sel != 0);
#pragma unroll
    for (int i = 0; i < 16; i++) {
        float av = fabsf(v[i]);
        float q = use_e1 ? q_e1m2(av / s_e1) * s_e1 : q_nvfp4(av / s_nv) * s_nv;
        wq[base + i] = (_Float16)copysignf(q, v[i]);
    }
}

// ---------------------------------------------------------------------------
// f16 WMMA GEMM: out[m,n] = sum_k A[m,k] * W[n,k] + bias[n]
// Block: 256 threads (8 waves), tile 128(M) x 128(N), K-chunk 32.
// Wave grid 4x2 -> each wave computes 32x64 via 2x4 v_wmma_f32_16x16x32_f16.
// Staging: GLOBAL_LOAD_ASYNC_TO_LDS_B128 into a triple-buffered LDS stage
// (2 chunks in flight, tracked by ASYNCcnt), one barrier per K-chunk.
// LDS tiles padded to stride 40 f16 for conflict-free ds_load_b128 frags.
// ---------------------------------------------------------------------------
#define LSTR 40
#define TILE_ELEMS (128 * LSTR)             // f16 elements per stage buffer
#define TILE_BYTES (TILE_ELEMS * 2)         // 10240 B

__global__ __launch_bounds__(256) void gemm_f16_wmma(const _Float16* __restrict__ A,
                                                     const _Float16* __restrict__ Bw,
                                                     const float* __restrict__ bias,
                                                     float* __restrict__ out,
                                                     int M, int N, int K) {
    __shared__ __align__(16) _Float16 sA[3 * TILE_ELEMS];
    __shared__ __align__(16) _Float16 sB[3 * TILE_ELEMS];

    const int tid  = threadIdx.x;
    const int n0   = blockIdx.x * 128;
    const int m0   = blockIdx.y * 128;
    const int row  = tid >> 1;       // 0..127 : tile row this thread stages
    const int half = tid & 1;        // which 16-f16 chunk of the 32-wide row
    const int lane = tid & 31;
    const int wave = tid >> 5;
    const int wm   = (wave & 3) * 32;  // wave's M offset in tile
    const int wn   = (wave >> 2) * 64; // wave's N offset in tile
    const int h    = lane >> 4;        // lane half (0: lanes 0-15, 1: 16-31)
    const int lr   = lane & 15;        // m/n index within 16

    v8f acc[2][4] = {};

    // Per-thread staging addresses (32 contiguous bytes per tile per chunk).
    const unsigned stageOff = (unsigned)((row * LSTR + half * 16) * 2);
    const unsigned ldsA0 = (unsigned)(uintptr_t)(void*)sA + stageOff;
    const unsigned ldsB0 = (unsigned)(uintptr_t)(void*)sB + stageOff;
    const unsigned long long gA0 =
        (unsigned long long)(uintptr_t)(A  + (size_t)(m0 + row) * K + half * 16);
    const unsigned long long gB0 =
        (unsigned long long)(uintptr_t)(Bw + (size_t)(n0 + row) * K + half * 16);

    const int nk = K / 32;

    // Issue async copy of K-chunk c into stage buffer b (4 ops -> ASYNCcnt+=4).
    // INST_OFFSET applies to both the LDS and global side, so the two 16B
    // pieces of the 32B slot share base registers.
    auto issue = [&](int c, int b) {
        unsigned la = ldsA0 + (unsigned)b * TILE_BYTES;
        unsigned lb = ldsB0 + (unsigned)b * TILE_BYTES;
        unsigned long long ga = gA0 + (unsigned long long)c * 64;
        unsigned long long gb = gB0 + (unsigned long long)c * 64;
        asm volatile("global_load_async_to_lds_b128 %0, %1, off"
                     :: "v"(la), "v"(ga) : "memory");
        asm volatile("global_load_async_to_lds_b128 %0, %1, off offset:16"
                     :: "v"(la), "v"(ga) : "memory");
        asm volatile("global_load_async_to_lds_b128 %0, %1, off"
                     :: "v"(lb), "v"(gb) : "memory");
        asm volatile("global_load_async_to_lds_b128 %0, %1, off offset:16"
                     :: "v"(lb), "v"(gb) : "memory");
    };

    issue(0, 0);
    if (nk > 1) issue(1, 1);

    for (int i = 0; i < nk; ++i) {
        // Chunk i complete when at most chunk i+1's 4 ops remain outstanding.
        if (i + 1 < nk) {
            asm volatile("s_wait_asynccnt 0x4" ::: "memory");
        } else {
            asm volatile("s_wait_asynccnt 0x0" ::: "memory");
        }
        __syncthreads();   // chunk i visible everywhere; compute(i-1) done in all waves

        // Refill: buffer (i+2)%3 == (i-1)%3, last read before the barrier above.
        if (i + 2 < nk) issue(i + 2, (i + 2) % 3);

        const _Float16* tA = sA + (i % 3) * TILE_ELEMS;
        const _Float16* tB = sB + (i % 3) * TILE_ELEMS;

        // Fragment loads: ISA 16-bit A/B layout — lane%16 selects row (m or n),
        // frag elems 0..7 = K[8h .. 8h+7], elems 8..15 = K[16+8h .. 23+8h].
        v16h af[2], bf[4];
#pragma unroll
        for (int mt = 0; mt < 2; mt++) {
            const _Float16* p = tA + (wm + mt * 16 + lr) * LSTR + 8 * h;
            v8h lo = *reinterpret_cast<const v8h*>(p);
            v8h hi = *reinterpret_cast<const v8h*>(p + 16);
            af[mt] = __builtin_shufflevector(lo, hi, 0,1,2,3,4,5,6,7,8,9,10,11,12,13,14,15);
        }
#pragma unroll
        for (int nt = 0; nt < 4; nt++) {
            const _Float16* p = tB + (wn + nt * 16 + lr) * LSTR + 8 * h;
            v8h lo = *reinterpret_cast<const v8h*>(p);
            v8h hi = *reinterpret_cast<const v8h*>(p + 16);
            bf[nt] = __builtin_shufflevector(lo, hi, 0,1,2,3,4,5,6,7,8,9,10,11,12,13,14,15);
        }
#pragma unroll
        for (int mt = 0; mt < 2; mt++)
#pragma unroll
            for (int nt = 0; nt < 4; nt++)
                acc[mt][nt] = __builtin_amdgcn_wmma_f32_16x16x32_f16(
                    false, af[mt], false, bf[nt], (short)0, acc[mt][nt], false, false);
    }

    // Epilogue: C layout — VGPR v: lanes 0-15 -> M = v, lanes 16-31 -> M = 8+v; N = lane%16.
#pragma unroll
    for (int mt = 0; mt < 2; mt++) {
#pragma unroll
        for (int nt = 0; nt < 4; nt++) {
            const int n = n0 + wn + nt * 16 + lr;
            const float bn = bias[n];
#pragma unroll
            for (int v = 0; v < 8; v++) {
                const int m = m0 + wm + mt * 16 + 8 * h + v;
                out[(size_t)m * N + n] = acc[mt][nt][v] + bn;
            }
        }
    }
}

// ---------------------------------------------------------------------------
// Inputs (setup_inputs order): x[f32], weight[f32], bias[f32], H_block[f32],
// signs[f32]. H_block is Sylvester/8 == analytic FWHT, so it is not read.
// ---------------------------------------------------------------------------
extern "C" void kernel_launch(void* const* d_in, const int* in_sizes, int n_in,
                              void* d_out, int out_size, void* d_ws, size_t ws_size,
                              hipStream_t stream) {
    const float* x     = (const float*)d_in[0];
    const float* wgt   = (const float*)d_in[1];
    const float* bias  = (const float*)d_in[2];
    const float* signs = (const float*)d_in[4];

    const int D_in  = in_sizes[4];            // 4096
    const int D_out = in_sizes[2];            // 4096
    const int M     = in_sizes[0] / D_in;     // B*T = 4096

    _Float16* xq = (_Float16*)d_ws;
    _Float16* wq = xq + (size_t)M * D_in;

    rotq_act<<<M, 256, 0, stream>>>(x, signs, xq, D_in);
    rotq_wgt<<<D_out, 256, 0, stream>>>(wgt, signs, wq, D_in);

    dim3 grid(D_out / 128, M / 128);
    gemm_f16_wmma<<<grid, 256, 0, stream>>>(xq, wq, bias, (float*)d_out,
                                            M, D_out, D_in);
}
